// WindowAttention3D_23441931502130
// MI455X (gfx1250) — matrix operand order repaired and checked by reading, not compiled, not run on hardware
//
#include <hip/hip_runtime.h>
#include <hip/hip_bf16.h>

typedef _Float16 h16;
typedef __attribute__((ext_vector_type(16))) _Float16 v16h;
typedef __attribute__((ext_vector_type(8)))  _Float16 v8h;
typedef __attribute__((ext_vector_type(8)))  float    v8f;
typedef __attribute__((ext_vector_type(4)))  unsigned int u32x4;
typedef __attribute__((ext_vector_type(8)))  int i32x8;
typedef __attribute__((ext_vector_type(4)))  int i32x4;

#define DIMC   256
#define HEADS  8
#define HD     32
#define NTOK   98
#define NPAD   112     // 7 M-tiles of 16
#define KPAD   128     // padded key dim for attn@V (4 WMMA K-steps)
#define BWIN   512
#define NWMASK 64
#define NN     (NTOK*NTOK)   // 9604
#define SPITCH 116           // LDS score pitch (floats)
#define APITCH 128           // LDS f16 attn pitch (halves)
#define SCALE_Q 0.17677669529663687f   // 32^-0.5

// LDS layout for attention kernel (bytes)
#define SMEM_S   0
#define SMEM_A16 ((size_t)NPAD * SPITCH * 4)                 // 51968
#define SMEM_KS  (SMEM_A16 + (size_t)NPAD * APITCH * 2)      // 80640
#define SMEM_VS  (SMEM_KS + (size_t)NPAD * HD * 2)           // 87808
#define SMEM_TOT (SMEM_VS + (size_t)HD * KPAD * 2)           // 96000

static __device__ __forceinline__ v16h cat8(v8h lo, v8h hi) {
  return __builtin_shufflevector(lo, hi, 0,1,2,3,4,5,6,7,8,9,10,11,12,13,14,15);
}
static __device__ __forceinline__ v8h  ld8 (const h16* p) { return *(const v8h*)p;  }
static __device__ __forceinline__ v16h ld16(const h16* p) { return *(const v16h*)p; }

// ---- Tensor Data Mover: 2D tile load global->LDS (D# per CDNA5 ISA ch.8) ----
// t0 = elements per row, t1 = rows, stride = row stride (f16 elements).
static __device__ __forceinline__ void tdm_load_2d(void* lds, const void* g,
                                                   unsigned t0, unsigned t1,
                                                   unsigned stride) {
  unsigned ldsa = (unsigned)(uintptr_t)lds;                 // LDS byte address (low 32)
  unsigned long long ga = (unsigned long long)(uintptr_t)g; // 57-bit global address
  u32x4 g0;
  g0[0] = 1u;                                               // count=1, user descriptor
  g0[1] = ldsa;                                             // lds_addr
  g0[2] = (unsigned)ga;                                     // global_addr[31:0]
  g0[3] = (unsigned)((ga >> 32) & 0x1FFFFFFull) | (2u << 30); // addr[56:32] | type=2
  i32x8 g1;
  g1[0] = 0x00010000;                                       // wg_mask=0, data_size=1 (2B)
  g1[1] = (int)((t0 & 0xFFFFu) << 16);                      // tensor_dim0[15:0]
  g1[2] = (int)(((t0 >> 16) & 0xFFFFu) | ((t1 & 0xFFFFu) << 16)); // dim0 hi | dim1 lo
  g1[3] = (int)(((t1 >> 16) & 0xFFFFu) | ((t0 & 0xFFFFu) << 16)); // dim1 hi | tile_dim0
  g1[4] = (int)(t1 & 0xFFFFu);                              // tile_dim1 (tile_dim2=0)
  g1[5] = (int)stride;                                      // tensor_dim0_stride[31:0]
  g1[6] = 0;                                                // stride hi | dim1_stride lo
  g1[7] = 0;
  i32x4 z4 = {0, 0, 0, 0};
#if __clang_major__ >= 23
  i32x8 z8 = {0, 0, 0, 0, 0, 0, 0, 0};
  __builtin_amdgcn_tensor_load_to_lds(g0, g1, z4, z4, z8, 0);
#else
  __builtin_amdgcn_tensor_load_to_lds(g0, g1, z4, z4, 0);
#endif
}

// ---------------- prep kernels ----------------

// dst[c*K + k] = (h16)src[k*C + c]   (src is K x C row-major, dst is C x K)
__global__ void transpose_to_h16(const float* __restrict__ src, h16* __restrict__ dst,
                                 int K, int C) {
  int idx = blockIdx.x * blockDim.x + threadIdx.x;
  if (idx >= K * C) return;
  int c = idx / K, kk = idx - c * K;
  dst[idx] = (h16)src[(size_t)kk * C + c];
}

// biasArr[h*NN + n*NTOK + m] = table[rpi[n*NTOK+m]*HEADS + h]
__global__ void build_bias(const float* __restrict__ table, const int* __restrict__ rpi,
                           float* __restrict__ biasArr) {
  int idx = blockIdx.x * blockDim.x + threadIdx.x;
  if (idx >= HEADS * NN) return;
  int h = idx / NN, nm = idx - h * NN;
  biasArr[idx] = table[(size_t)rpi[nm] * HEADS + h];
}

// ---------------- kernel 1: QKV GEMM (x @ qkv_w + b) ----------------
// grid: (512*7, 24), block: 32 (one wave). Tile: 16 tokens x 32 cols, K=256.
__global__ void __launch_bounds__(32)
qkv_gemm(const float* __restrict__ x, const h16* __restrict__ wT,
         const float* __restrict__ qkv_b,
         h16* __restrict__ qws, h16* __restrict__ kws, h16* __restrict__ vt) {
  int lane = threadIdx.x;
  int ln = lane & 15, hig = lane >> 4;
  int hi8 = hig * 8, koff = hig * 16;
  int b = blockIdx.x / 7, t = blockIdx.x % 7;
  int n0 = blockIdx.y * 32;
  int s = n0 >> 8, h = (n0 >> 5) & 7;

  // prefetch this wave's weight rows (global_prefetch_b8)
  __builtin_prefetch(wT + (size_t)(n0 + ln) * DIMC, 0, 3);
  __builtin_prefetch(wT + (size_t)(n0 + 16 + ln) * DIMC, 0, 3);

  int token = t * 16 + ln; if (token > NTOK - 1) token = NTOK - 1;  // clamp pad rows
  const float* xrow = x + (size_t)(b * NTOK + token) * DIMC;

  v8f acc0 = {}, acc1 = {};
  for (int kb = 0; kb < DIMC; kb += 32) {
    v16h A;
    const float* p0 = xrow + kb + koff;
    const float* p1 = xrow + kb + koff + 8;
#pragma unroll
    for (int e = 0; e < 8; ++e) { A[e] = (h16)p0[e]; A[e + 8] = (h16)p1[e]; }

    v16h B0 = ld16(wT + (size_t)(n0 + ln) * DIMC + kb + koff);
    acc0 = __builtin_amdgcn_wmma_f32_16x16x32_f16(false, A, false, B0, (short)0, acc0, false, false);
    v16h B1 = ld16(wT + (size_t)(n0 + 16 + ln) * DIMC + kb + koff);
    acc1 = __builtin_amdgcn_wmma_f32_16x16x32_f16(false, A, false, B1, (short)0, acc1, false, false);
  }
#pragma unroll
  for (int j = 0; j < 2; ++j) {
    v8f acc = j ? acc1 : acc0;
    int c = n0 + j * 16 + ln;
    float bias = qkv_b[c];
    int d = c & 31;
#pragma unroll
    for (int r = 0; r < 8; ++r) {
      int tok = t * 16 + r + hi8;               // <= 111, buffers padded to NPAD
      float val = acc[r] + bias;
      if (s == 0) {
        qws[((size_t)(b * HEADS + h) * NPAD + tok) * HD + d] = (h16)(val * SCALE_Q);
      } else if (s == 1) {
        kws[((size_t)(b * HEADS + h) * NPAD + tok) * HD + d] = (h16)val;
      } else if (tok < NTOK) {                  // keep zero pad for keys >= 98
        vt[((size_t)(b * HEADS + h) * HD + d) * KPAD + tok] = (h16)val;
      }
    }
  }
}

// ---------------- kernel 2: attention per (b, h) ----------------
// grid: 4096, block: 128 (4 waves).
// LDS: S[112][116] f32, A16[112][128] f16, Ks[112][32] f16, Vs[32][128] f16.
__global__ void __launch_bounds__(128)
attn_kernel(const h16* __restrict__ qws, const h16* __restrict__ kws,
            const h16* __restrict__ vt, const float* __restrict__ biasArr,
            const float* __restrict__ mask, h16* __restrict__ attn_out) {
  extern __shared__ char smem[];
  float* S  = (float*)(smem + SMEM_S);
  h16*  A16 = (h16*)(smem + SMEM_A16);
  h16*  Ks  = (h16*)(smem + SMEM_KS);
  h16*  Vs  = (h16*)(smem + SMEM_VS);

  int tid = threadIdx.x;
  int wid = tid >> 5, lane = tid & 31;
  int ln = lane & 15, hig = lane >> 4;
  int hi8 = hig * 8, koff = hig * 16;
  int b = blockIdx.x >> 3, h = blockIdx.x & 7;

  const h16* qbase = qws + (size_t)(b * HEADS + h) * NPAD * HD;
  const h16* kbase = kws + (size_t)(b * HEADS + h) * NPAD * HD;
  const h16* vbase = vt  + (size_t)(b * HEADS + h) * HD * KPAD;

  // ---- TDM: DMA the K-slab and V^T-slab into LDS (wave 0 issues) ----
  if (wid == 0) {
    tdm_load_2d(Ks, kbase, HD, NPAD, HD);      // 112 x 32 f16, contiguous
    tdm_load_2d(Vs, vbase, KPAD, HD, KPAD);    // 32 x 128 f16, contiguous
    __builtin_amdgcn_s_wait_tensorcnt(0);
  }
  __syncthreads();

  // ---- phase 1: S = (q*scale) @ k^T  (49 WMMA tiles split over 4 waves) ----
  for (int mt = wid; mt < 7; mt += 4) {
    const h16* qrow = qbase + (size_t)(mt * 16 + ln) * HD;
    v16h A = cat8(ld8(qrow + koff), ld8(qrow + koff + 8));
    for (int nt = 0; nt < 7; ++nt) {
      v16h B = ld16(Ks + (size_t)(nt * 16 + ln) * HD + koff);
      v8f acc = {};
      acc = __builtin_amdgcn_wmma_f32_16x16x32_f16(false, A, false, B, (short)0, acc, false, false);
      int col = nt * 16 + ln;
#pragma unroll
      for (int r = 0; r < 8; ++r)
        S[(mt * 16 + r + hi8) * SPITCH + col] = acc[r];
    }
  }
  __syncthreads();

  // ---- phase 2: +bias +mask, softmax over keys, emit f16 probs ----
  if (tid < NPAD) {
    int n = tid;
    if (n < NTOK) {
      const float* brow = biasArr + (size_t)h * NN + (size_t)n * NTOK;
      const float* mrow = mask + (size_t)(b & (NWMASK - 1)) * NN + (size_t)n * NTOK;
      float mx = -1e30f;
      for (int m = 0; m < NTOK; ++m) {
        float sv = S[n * SPITCH + m] + brow[m] + mrow[m];
        S[n * SPITCH + m] = sv;
        mx = fmaxf(mx, sv);
      }
      float sum = 0.f;
      for (int m = 0; m < NTOK; ++m) {
        float e = __expf(S[n * SPITCH + m] - mx);
        S[n * SPITCH + m] = e;
        sum += e;
      }
      float inv = 1.f / sum;
      for (int m = 0; m < NTOK; ++m)
        A16[n * APITCH + m] = (h16)(S[n * SPITCH + m] * inv);
      for (int m = NTOK; m < KPAD; ++m)
        A16[n * APITCH + m] = (h16)0.f;
    } else {
      for (int m = 0; m < KPAD; ++m) A16[n * APITCH + m] = (h16)0.f;  // pad rows
    }
  }
  __syncthreads();

  // ---- phase 3: out = attn @ v  (K=128 padded, B from LDS V^T) ----
  for (int mt = wid; mt < 7; mt += 4) {
    v8f acc0 = {}, acc1 = {};
    const h16* arow = A16 + (size_t)(mt * 16 + ln) * APITCH;
    for (int kb = 0; kb < KPAD; kb += 32) {
      v16h A = cat8(*(const v8h*)(arow + kb + koff), *(const v8h*)(arow + kb + koff + 8));
      v16h B0 = ld16(Vs + (size_t)ln * KPAD + kb + koff);
      acc0 = __builtin_amdgcn_wmma_f32_16x16x32_f16(false, A, false, B0, (short)0, acc0, false, false);
      v16h B1 = ld16(Vs + (size_t)(16 + ln) * KPAD + kb + koff);
      acc1 = __builtin_amdgcn_wmma_f32_16x16x32_f16(false, A, false, B1, (short)0, acc1, false, false);
    }
#pragma unroll
    for (int r = 0; r < 8; ++r) {
      int row = mt * 16 + r + hi8;
      attn_out[((size_t)b * NPAD + row) * DIMC + h * HD + ln]      = (h16)acc0[r];
      attn_out[((size_t)b * NPAD + row) * DIMC + h * HD + 16 + ln] = (h16)acc1[r];
    }
  }
}

// ---------------- kernel 3: output projection ----------------
// grid: (512*7, 8), block: 32. Tile: 16 rows x 32 cols, K=256.
__global__ void __launch_bounds__(32)
proj_gemm(const h16* __restrict__ ain, const h16* __restrict__ pT,
          const float* __restrict__ pb, float* __restrict__ out) {
  int lane = threadIdx.x;
  int ln = lane & 15, hig = lane >> 4;
  int hi8 = hig * 8, koff = hig * 16;
  int b = blockIdx.x / 7, t = blockIdx.x % 7;
  int n0 = blockIdx.y * 32;

  __builtin_prefetch(pT + (size_t)(n0 + ln) * DIMC, 0, 3);
  __builtin_prefetch(pT + (size_t)(n0 + 16 + ln) * DIMC, 0, 3);

  const h16* arow = ain + ((size_t)b * NPAD + t * 16 + ln) * DIMC;
  v8f acc0 = {}, acc1 = {};
  for (int kb = 0; kb < DIMC; kb += 32) {
    v16h A = cat8(ld8(arow + kb + koff), ld8(arow + kb + koff + 8));
    v16h B0 = ld16(pT + (size_t)(n0 + ln) * DIMC + kb + koff);
    acc0 = __builtin_amdgcn_wmma_f32_16x16x32_f16(false, A, false, B0, (short)0, acc0, false, false);
    v16h B1 = ld16(pT + (size_t)(n0 + 16 + ln) * DIMC + kb + koff);
    acc1 = __builtin_amdgcn_wmma_f32_16x16x32_f16(false, A, false, B1, (short)0, acc1, false, false);
  }
#pragma unroll
  for (int j = 0; j < 2; ++j) {
    v8f acc = j ? acc1 : acc0;
    int c = n0 + j * 16 + ln;
    float bias = pb[c];
#pragma unroll
    for (int r = 0; r < 8; ++r) {
      int tok = t * 16 + r + hi8;
      if (tok < NTOK)
        out[((size_t)b * NTOK + tok) * DIMC + c] = acc[r] + bias;
    }
  }
}

// ---------------- host launcher ----------------
extern "C" void kernel_launch(void* const* d_in, const int* in_sizes, int n_in,
                              void* d_out, int out_size, void* d_ws, size_t ws_size,
                              hipStream_t stream) {
  const float* x       = (const float*)d_in[0];
  const float* qkv_w   = (const float*)d_in[1];
  const float* qkv_b   = (const float*)d_in[2];
  const float* btable  = (const float*)d_in[3];
  const float* proj_w  = (const float*)d_in[4];
  const float* proj_b  = (const float*)d_in[5];
  const float* mask    = (const float*)d_in[6];
  const int*   rpi     = (const int*)  d_in[7];
  float* outp = (float*)d_out;

  char* ws = (char*)d_ws;
  size_t off = 0;
  auto take = [&](size_t bytes) -> void* {
    void* p = ws + off;
    off += (bytes + 255) & ~(size_t)255;
    return p;
  };
  h16*   wT      = (h16*)  take((size_t)768 * 256 * 2);           // qkv_w^T f16
  h16*   pT      = (h16*)  take((size_t)256 * 256 * 2);           // proj_w^T f16
  float* biasArr = (float*)take((size_t)HEADS * NN * 4);          // [h][n][m]
  h16*   qws     = (h16*)  take((size_t)BWIN * HEADS * NPAD * HD * 2);
  h16*   kws     = (h16*)  take((size_t)BWIN * HEADS * NPAD * HD * 2);
  size_t vt_bytes = (size_t)BWIN * HEADS * HD * KPAD * 2;
  h16*   vt      = (h16*)  take(vt_bytes);
  h16*   aout    = (h16*)  take((size_t)BWIN * NPAD * DIMC * 2);
  (void)ws_size; (void)in_sizes; (void)n_in; (void)out_size;

  // prep
  transpose_to_h16<<<(768 * 256 + 255) / 256, 256, 0, stream>>>(qkv_w, wT, 256, 768);
  transpose_to_h16<<<(256 * 256 + 255) / 256, 256, 0, stream>>>(proj_w, pT, 256, 256);
  build_bias<<<(HEADS * NN + 255) / 256, 256, 0, stream>>>(btable, rpi, biasArr);
  hipMemsetAsync(vt, 0, vt_bytes, stream);   // zero pad keys 98..127

  // pipeline
  qkv_gemm<<<dim3(BWIN * 7, 24), 32, 0, stream>>>(x, wT, qkv_b, qws, kws, vt);

  attn_kernel<<<dim3(BWIN * HEADS), 128, SMEM_TOT, stream>>>(qws, kws, vt, biasArr, mask, aout);

  proj_gemm<<<dim3(BWIN * 7, 8), 32, 0, stream>>>(aout, pT, proj_b, outp);
}
